// EncoderLayer_12567074308450
// MI455X (gfx1250) — matrix operand order repaired and checked by reading, not compiled, not run on hardware
//
#include <hip/hip_runtime.h>
#include <hip/hip_bf16.h>

typedef __attribute__((ext_vector_type(16))) _Float16 v16h;
typedef __attribute__((ext_vector_type(4)))  _Float16 v4h;
typedef __attribute__((ext_vector_type(8)))  float    v8f;
typedef __attribute__((ext_vector_type(4)))  float    v4f;

// ---- problem constants (from reference) ----
static constexpr int BB   = 4;
static constexpr int TT   = 1024;
static constexpr int DD   = 512;
static constexpr int HH   = 8;
static constexpr int DHH  = 64;     // D / H
static constexpr int DFFC = 2048;
static constexpr int EE   = 8;
static constexpr int NTOK = BB * TT;
static constexpr float EPSC = 1e-5f;

// =====================================================================
// Batched GEMM: C[bz] = epilogue( alpha * A[bz] @ B[bz] + bias )
// fp32 in/out, f16 WMMA 16x16x32 inside, f32 accumulation.
// Block tile BM x BN, 8 waves, wave tile (16*FM) x (16*FN).
// ALL dims must be tile-aligned (every call here is) -> no bounds checks.
// =====================================================================
template<int BM, int BN, int FM, int FN, int TRANSB>
__global__ __launch_bounds__(256) void gemm_wmma(
    const float* __restrict__ A, int lda, long long strideA,
    const float* __restrict__ Bm, int ldb, long long strideB,
    const float* __restrict__ bias,
    const float* __restrict__ rowScale, int rsStride,
    float* __restrict__ C, int ldc, long long strideC,
    int K, float alpha, int doRelu, int accum)
{
  constexpr int WAVES_M = BM / (16 * FM);
  constexpr int WAVES_N = BN / (16 * FN);
  static_assert(WAVES_M * WAVES_N == 8, "8 waves per block");

  const int bz = blockIdx.z;
  A  += (long long)bz * strideA;
  Bm += (long long)bz * strideB;
  C  += (long long)bz * strideC;

  const int m0 = blockIdx.y * BM;
  const int n0 = blockIdx.x * BN;
  const int tid  = threadIdx.x;
  const int wave = tid >> 5;
  const int lane = tid & 31;
  const int waveM = wave / WAVES_N;
  const int waveN = wave % WAVES_N;
  const int kh = lane >> 4;                // lane half (ISA fragment split)
  const int lm = lane & 15;

  __shared__ __align__(16) _Float16 sA[BM * 40];   // [row][k], pad 40
  __shared__ __align__(16) _Float16 sB[BN * 40];   // [col][k] (K-transposed), pad 40

  v8f acc[FM][FN] = {};

  for (int k0 = 0; k0 < K; k0 += 32) {
    // ---- stage A tile (BM x 32): float4 loads, packed cvt, b64 LDS stores ----
#pragma unroll
    for (int i = 0; i < BM / 32; ++i) {
      int c = tid + i * 256;               // 8 float4-chunks per row
      int r = c >> 3, kk = (c & 7) * 4;
      const float* ap = &A[(long long)(m0 + r) * lda + (k0 + kk)];
      v4f fv = *(const v4f*)ap;
      __builtin_prefetch(ap + 32, 0, 1);   // speculative, next K tile
      *(v4h*)&sA[r * 40 + kk] = __builtin_convertvector(fv, v4h);
    }
    // ---- stage B tile (32K x BN) into sB[n][k] ----
#pragma unroll
    for (int i = 0; i < BN / 32; ++i) {
      int c = tid + i * 256;
      if (TRANSB) {                        // B[n][k]: contiguous along k
        int n = c >> 3, kk = (c & 7) * 4;
        v4f fv = *(const v4f*)&Bm[(long long)(n0 + n) * ldb + (k0 + kk)];
        *(v4h*)&sB[n * 40 + kk] = __builtin_convertvector(fv, v4h);
      } else {                             // B[k][n]: 4 coalesced b32 down K, pack, one b64 store
        int n = c % BN, kk = (c / BN) * 4;
        const float* bp = &Bm[(long long)(k0 + kk) * ldb + (n0 + n)];
        v4f fv;
        fv[0] = bp[0];
        fv[1] = bp[(long long)ldb];
        fv[2] = bp[(long long)ldb * 2];
        fv[3] = bp[(long long)ldb * 3];
        *(v4h*)&sB[n * 40 + kk] = __builtin_convertvector(fv, v4h);
      }
    }
    __syncthreads();

    // ---- fragments per ISA VGPR layout ----
    v16h af[FM], bf[FN];
    const int ra = waveM * 16 * FM;
    const int rb = waveN * 16 * FN;
#pragma unroll
    for (int p = 0; p < FM; ++p) {
#pragma unroll
      for (int j = 0; j < 8; ++j) {
        int bk = ((j < 4) ? 0 : 16) + kh * 8 + (j & 3) * 2;   // A: K pairs
        af[p][2 * j]     = sA[(ra + p * 16 + lm) * 40 + bk];
        af[p][2 * j + 1] = sA[(ra + p * 16 + lm) * 40 + bk + 1];
      }
    }
#pragma unroll
    for (int q = 0; q < FN; ++q) {
#pragma unroll
      for (int j = 0; j < 8; ++j) {
        int bk = ((j < 4) ? 0 : 16) + kh * 8 + (j & 3) * 2;   // B: K pairs
        bf[q][2 * j]     = sB[(rb + q * 16 + lm) * 40 + bk];
        bf[q][2 * j + 1] = sB[(rb + q * 16 + lm) * 40 + bk + 1];
      }
    }
#pragma unroll
    for (int p = 0; p < FM; ++p)
#pragma unroll
      for (int q = 0; q < FN; ++q)
        acc[p][q] = __builtin_amdgcn_wmma_f32_16x16x32_f16(
            false, af[p], false, bf[q], (short)0, acc[p][q], false, false);
    __syncthreads();
  }

  // ---- epilogue ----
  const int mwave = m0 + waveM * 16 * FM;
  const int nwave = n0 + waveN * 16 * FN;
#pragma unroll
  for (int p = 0; p < FM; ++p) {
#pragma unroll
    for (int q = 0; q < FN; ++q) {
      int col = nwave + q * 16 + lm;
      float bval = bias ? bias[col] : 0.f;
#pragma unroll
      for (int r = 0; r < 8; ++r) {
        int row = mwave + p * 16 + kh * 8 + r;   // C/D layout: VGPR r -> M = r + 8*laneHalf
        float v = acc[p][q][r] * alpha + bval;
        if (doRelu) v = v > 0.f ? v : 0.f;
        if (rowScale) v *= rowScale[(long long)row * rsStride];
        long long off = (long long)row * ldc + col;
        if (accum) C[off] += v; else C[off] = v;
      }
    }
  }
}

// =====================================================================
// [B,T,H*DH] <-> [B,H,T,DH] reorder. dir=0: btd->bhtd, dir=1: bhtd->btd
// =====================================================================
__global__ void reorder_kernel(const float* __restrict__ in, float* __restrict__ out, int dir) {
  long long i = (long long)blockIdx.x * blockDim.x + threadIdx.x;
  if (i >= (long long)NTOK * DD) return;
  int b = (int)(i / ((long long)TT * DD));
  int rem = (int)(i % ((long long)TT * DD));
  int t = rem / DD, d = rem % DD;
  int h = d / DHH, dh = d % DHH;
  long long bhtd = (((long long)b * HH + h) * TT + t) * DHH + dh;
  if (dir == 0) out[bhtd] = in[i];
  else          out[i]    = in[bhtd];
}

// ===================== rowwise softmax, row length 1024 ==============
__global__ __launch_bounds__(256) void softmax_rows(float* __restrict__ data) {
  float* p = data + (long long)blockIdx.x * 1024;
  const int tid = threadIdx.x, wave = tid >> 5, lane = tid & 31;
  __shared__ float red[8];
  float vals[4], mx = -3.0e38f;
#pragma unroll
  for (int j = 0; j < 4; ++j) { vals[j] = p[tid + j * 256]; mx = fmaxf(mx, vals[j]); }
  for (int off = 16; off >= 1; off >>= 1) mx = fmaxf(mx, __shfl_xor(mx, off, 32));
  if (lane == 0) red[wave] = mx;
  __syncthreads();
  float m2 = red[0];
#pragma unroll
  for (int w = 1; w < 8; ++w) m2 = fmaxf(m2, red[w]);
  __syncthreads();
  float s = 0.f;
#pragma unroll
  for (int j = 0; j < 4; ++j) { vals[j] = __expf(vals[j] - m2); s += vals[j]; }
  for (int off = 16; off >= 1; off >>= 1) s += __shfl_xor(s, off, 32);
  if (lane == 0) red[wave] = s;
  __syncthreads();
  float tot = 0.f;
#pragma unroll
  for (int w = 0; w < 8; ++w) tot += red[w];
  float inv = 1.f / tot;
#pragma unroll
  for (int j = 0; j < 4; ++j) p[tid + j * 256] = vals[j] * inv;
}

// ============ out = LayerNorm(a + b) * g + beta, row length 512 =======
__global__ __launch_bounds__(256) void add_ln(const float* __restrict__ a, const float* __restrict__ b,
                                              const float* __restrict__ g, const float* __restrict__ be,
                                              float* __restrict__ out) {
  long long row = blockIdx.x;
  const int tid = threadIdx.x, wave = tid >> 5, lane = tid & 31;
  __shared__ float red[8];
  long long base = row * DD;
  float z0 = a[base + tid]       + b[base + tid];
  float z1 = a[base + tid + 256] + b[base + tid + 256];
  float s = z0 + z1;
  for (int off = 16; off >= 1; off >>= 1) s += __shfl_xor(s, off, 32);
  if (lane == 0) red[wave] = s;
  __syncthreads();
  float tot = 0.f;
#pragma unroll
  for (int w = 0; w < 8; ++w) tot += red[w];
  float mean = tot * (1.f / DD);
  __syncthreads();
  float d0 = z0 - mean, d1 = z1 - mean;
  float v = d0 * d0 + d1 * d1;
  for (int off = 16; off >= 1; off >>= 1) v += __shfl_xor(v, off, 32);
  if (lane == 0) red[wave] = v;
  __syncthreads();
  float vt = 0.f;
#pragma unroll
  for (int w = 0; w < 8; ++w) vt += red[w];
  float rs = rsqrtf(vt * (1.f / DD) + EPSC);
  out[base + tid]       = d0 * rs * g[tid]       + be[tid];
  out[base + tid + 256] = d1 * rs * g[tid + 256] + be[tid + 256];
}

// ============== gate logits: [NTOK,E] = x1 @ Wg (D x E, E=8) ==========
__global__ void gate_logits_kernel(const float* __restrict__ x1, const float* __restrict__ Wg,
                                   float* __restrict__ logits) {
  int i = blockIdx.x * blockDim.x + threadIdx.x;
  if (i >= NTOK * EE) return;
  int m = i >> 3, e = i & 7;
  const float* xr = x1 + (long long)m * DD;
  float s = 0.f;
  for (int d = 0; d < DD; ++d) s += xr[d] * Wg[(long long)d * EE + e];
  logits[i] = s;
}

// ========= softmax over E=8 + top-2 combine weights per token =========
__global__ void gate_topk_kernel(const float* __restrict__ logits, float* __restrict__ gate_out,
                                 float* __restrict__ cw) {
  int m = blockIdx.x * blockDim.x + threadIdx.x;
  if (m >= NTOK) return;
  float g[EE], mx = -3.0e38f;
#pragma unroll
  for (int e = 0; e < EE; ++e) { g[e] = logits[m * EE + e]; mx = fmaxf(mx, g[e]); }
  float s = 0.f;
#pragma unroll
  for (int e = 0; e < EE; ++e) { g[e] = __expf(g[e] - mx); s += g[e]; }
  float inv = 1.f / s;
#pragma unroll
  for (int e = 0; e < EE; ++e) { g[e] *= inv; gate_out[m * EE + e] = g[e]; }
  int i1 = 0;
#pragma unroll
  for (int e = 1; e < EE; ++e) if (g[e] > g[i1]) i1 = e;       // first max on ties
  int i2 = (i1 == 0) ? 1 : 0;
#pragma unroll
  for (int e = 0; e < EE; ++e) if (e != i1 && g[e] > g[i2]) i2 = e;
#pragma unroll
  for (int e = 0; e < EE; ++e)
    cw[m * EE + e] = (e == i1) ? g[i1] : ((e == i2) ? g[i2] : 0.f);
}

__global__ void zero_kernel(float* __restrict__ p, long long n) {
  long long i = (long long)blockIdx.x * blockDim.x + threadIdx.x;
  if (i < n) p[i] = 0.f;
}

// =====================================================================
extern "C" void kernel_launch(void* const* d_in, const int* in_sizes, int n_in,
                              void* d_out, int out_size, void* d_ws, size_t ws_size,
                              hipStream_t stream) {
  (void)in_sizes; (void)n_in; (void)out_size; (void)ws_size;
  const float* x   = (const float*)d_in[0];
  const float* Wq  = (const float*)d_in[1];
  const float* bq  = (const float*)d_in[2];
  const float* Wk  = (const float*)d_in[3];
  const float* bk  = (const float*)d_in[4];
  const float* Wv  = (const float*)d_in[5];
  const float* bv  = (const float*)d_in[6];
  const float* Wo  = (const float*)d_in[7];
  const float* bo  = (const float*)d_in[8];
  const float* g1  = (const float*)d_in[9];
  const float* be1 = (const float*)d_in[10];
  const float* g2  = (const float*)d_in[11];
  const float* be2 = (const float*)d_in[12];
  const float* Wg  = (const float*)d_in[13];
  const float* W1  = (const float*)d_in[14];
  const float* B1  = (const float*)d_in[15];
  const float* W2  = (const float*)d_in[16];
  const float* B2  = (const float*)d_in[17];

  float* outp = (float*)d_out;
  float* y_out    = outp;                                       // [B,T,D]
  float* attn_out = outp + (long long)NTOK * DD;                // [B,H,T,T]
  float* gate_out = attn_out + (long long)BB * HH * TT * TT;    // [B,T,E]

  float* ws = (float*)d_ws;
  long long off = 0;
  float* qb     = ws + off; off += (long long)NTOK * DD;    // [B,H,T,DH]
  float* kb     = ws + off; off += (long long)NTOK * DD;
  float* vb     = ws + off; off += (long long)NTOK * DD;
  float* ctx    = ws + off; off += (long long)NTOK * DD;    // [B,H,T,DH]
  float* tmp    = ws + off; off += (long long)NTOK * DD;    // [B,T,D] scratch
  float* nx     = ws + off; off += (long long)NTOK * DD;    // attn out-proj
  float* x1     = ws + off; off += (long long)NTOK * DD;    // post-LN1
  float* moe    = ws + off; off += (long long)NTOK * DD;    // MoE accumulator
  float* hbuf   = ws + off; off += (long long)NTOK * DFFC;  // expert hidden
  float* logits = ws + off; off += (long long)NTOK * EE;
  float* cw     = ws + off; off += (long long)NTOK * EE;

  const dim3 blk(256);
  const dim3 gProj(DD / 256, NTOK / 128, 1);       // 2 x 32 (128x256 tiles)
  const dim3 gElem((NTOK * DD + 255) / 256);

  // ---- Q, K, V projections (BTD) then reorder to BHTD ----
  gemm_wmma<128, 256, 4, 4, 0><<<gProj, blk, 0, stream>>>(x, DD, 0, Wq, DD, 0, bq, nullptr, 0,
                                                          tmp, DD, 0, DD, 1.f, 0, 0);
  reorder_kernel<<<gElem, blk, 0, stream>>>(tmp, qb, 0);
  gemm_wmma<128, 256, 4, 4, 0><<<gProj, blk, 0, stream>>>(x, DD, 0, Wk, DD, 0, bk, nullptr, 0,
                                                          tmp, DD, 0, DD, 1.f, 0, 0);
  reorder_kernel<<<gElem, blk, 0, stream>>>(tmp, kb, 0);
  gemm_wmma<128, 256, 4, 4, 0><<<gProj, blk, 0, stream>>>(x, DD, 0, Wv, DD, 0, bv, nullptr, 0,
                                                          tmp, DD, 0, DD, 1.f, 0, 0);
  reorder_kernel<<<gElem, blk, 0, stream>>>(tmp, vb, 0);

  // ---- scores = q @ k^T / sqrt(DH), batched over 32 (b,h); B transposed ----
  gemm_wmma<128, 256, 4, 4, 1><<<dim3(TT / 256, TT / 128, BB * HH), blk, 0, stream>>>(
      qb, DHH, (long long)TT * DHH, kb, DHH, (long long)TT * DHH,
      nullptr, nullptr, 0, attn_out, TT, (long long)TT * TT,
      DHH, 0.125f, 0, 0);

  // ---- softmax rows (this is also the attn output) ----
  softmax_rows<<<dim3(BB * HH * TT), blk, 0, stream>>>(attn_out);

  // ---- ctx = attn @ v (N=64 -> 128x64 tile, 2x2 wave frags), batched ----
  gemm_wmma<128, 64, 2, 2, 0><<<dim3(1, TT / 128, BB * HH), blk, 0, stream>>>(
      attn_out, TT, (long long)TT * TT, vb, DHH, (long long)TT * DHH,
      nullptr, nullptr, 0, ctx, DHH, (long long)TT * DHH,
      TT, 1.f, 0, 0);

  // ---- merge heads, output projection ----
  reorder_kernel<<<gElem, blk, 0, stream>>>(ctx, tmp, 1);
  gemm_wmma<128, 256, 4, 4, 0><<<gProj, blk, 0, stream>>>(tmp, DD, 0, Wo, DD, 0, bo, nullptr, 0,
                                                          nx, DD, 0, DD, 1.f, 0, 0);

  // ---- x1 = LN(x + new_x) ----
  add_ln<<<dim3(NTOK), blk, 0, stream>>>(x, nx, g1, be1, x1);

  // ---- gate softmax + top-2 combine weights ----
  gate_logits_kernel<<<dim3((NTOK * EE + 255) / 256), blk, 0, stream>>>(x1, Wg, logits);
  gate_topk_kernel<<<dim3((NTOK + 255) / 256), blk, 0, stream>>>(logits, gate_out, cw);

  // ---- MoE: out += cw[:,e] * (relu(x1@W1[e]+B1[e]) @ W2[e] + B2[e]) ----
  zero_kernel<<<gElem, blk, 0, stream>>>(moe, (long long)NTOK * DD);
  for (int e = 0; e < EE; ++e) {
    gemm_wmma<128, 256, 4, 4, 0><<<dim3(DFFC / 256, NTOK / 128, 1), blk, 0, stream>>>(
        x1, DD, 0, W1 + (long long)e * DD * DFFC, DFFC, 0,
        B1 + (long long)e * DFFC, nullptr, 0,
        hbuf, DFFC, 0, DD, 1.f, /*relu=*/1, 0);
    gemm_wmma<128, 256, 4, 4, 0><<<dim3(DD / 256, NTOK / 128, 1), blk, 0, stream>>>(
        hbuf, DFFC, 0, W2 + (long long)e * DFFC * DD, DD, 0,
        B2 + (long long)e * DD, cw + e, EE,
        moe, DD, 0, DFFC, 1.f, 0, /*accum=*/1);
  }

  // ---- y = LN(x1 + moe_out) ----
  add_ln<<<dim3(NTOK), blk, 0, stream>>>(x1, moe, g2, be2, y_out);
}
